// PLELayer_60120952209970
// MI455X (gfx1250) — matrix operand order repaired and checked by reading, not compiled, not run on hardware
//
#include <hip/hip_runtime.h>
#include <math.h>

typedef __attribute__((ext_vector_type(16))) _Float16 v16h;
typedef __attribute__((ext_vector_type(8)))  float    v8f;
typedef __attribute__((ext_vector_type(2)))  _Float16 h2v;

#define DEV static __device__ __forceinline__

// ---------------------------------------------------------------- constants
constexpr int BATCH  = 131072;
constexpr int NTILES = BATCH / 16;   // 8192 tiles of 16 rows

// workspace layout in halves (f16)
constexpr size_t OFF_W1F0 = 0;                      // 16 units * 3c * 4nt * 512
constexpr size_t OFF_W2F0 = OFF_W1F0 + 16*12*512;   // 16 units * 2c * 512
constexpr size_t OFF_W3F0 = OFF_W2F0 + 16*2*512;    // 4 units * 512
constexpr size_t OFF_W1F1 = OFF_W3F0 + 4*512;       // 15 units * 4nt * 512
constexpr size_t OFF_W2F1 = OFF_W1F1 + 15*4*512;    // 15 units * 2c * 512
constexpr size_t OFF_W3F1 = OFF_W2F1 + 15*2*512;    // 3 units * 512

// ---------------------------------------------------------------- helpers
DEV v8f wmma_f16(v16h a, v16h b, v8f c) {
  // D = A(16x32 f16) * B(32x16 f16) + C(16x16 f32)
  return __builtin_amdgcn_wmma_f32_16x16x32_f16(false, a, false, b, (short)0, c,
                                                false, false);
}

// Build A operand from an LDS row-major f16 tile [16][ld].
// 16-bit A layout: lane m=L&15, hi=L>>4; vgpr v holds K = 16*(v>=4)+8*hi+2*(v&3)+{0,1}
DEV v16h load_afrag(const _Float16* tile, int ld, int kbase, int lane) {
  const int m = lane & 15, hi = lane >> 4;
  const _Float16* row = tile + m * ld + kbase + hi * 8;
  v16h a;
#pragma unroll
  for (int v = 0; v < 8; ++v) {
    const int k = ((v & 4) ? 16 : 0) + (v & 3) * 2;
    h2v p = *(const h2v*)(row + k);
    a[2 * v] = p[0];
    a[2 * v + 1] = p[1];
  }
  return a;
}

// B operand pre-swizzled in global memory: frag[fragIdx][lane][16 halves]
DEV v16h load_bfrag(const _Float16* base, int fragIdx, int lane) {
  return ((const v16h*)base)[fragIdx * 32 + lane];
}

// ---------------------------------------------------------------- prep kernel
// Pack a row-major [K][N] f32 weight (possibly `units` of them) into the
// B-operand register image: K = 32c + 16*(h>>3) + 8*(lane>>4) + (h&7), N = 16*nt + (lane&15).
__global__ void pack_bfrag(const float* __restrict__ W, _Float16* __restrict__ dst,
                           int K, int N, int nC, int nT, int units) {
  const int per = nC * nT * 512;
  const int total = per * units;
  for (int i = blockIdx.x * blockDim.x + threadIdx.x; i < total;
       i += gridDim.x * blockDim.x) {
    const int u = i / per, j = i % per;
    const int h = j & 15, lane = (j >> 4) & 31, t2 = j >> 9;
    const int nt = t2 % nT, c = t2 / nT;
    const int k = c * 32 + ((h >> 3) << 4) + ((lane >> 4) << 3) + (h & 7);
    const int n = nt * 16 + (lane & 15);
    float v = (k < K && n < N) ? W[(size_t)u * K * N + (size_t)k * N + n] : 0.0f;
    dst[i] = (_Float16)v;
  }
}

// ---------------------------------------------------------------- main kernel
struct PleParams {
  const int* cat_idx;
  const float* emb;
  const float *spec0_b1, *shar0_b1, *gate0_b1, *sgate0_b1;
  const float *spec0_b2, *shar0_b2, *gate0_b2, *sgate0_b2;
  const float *spec1_b1, *shar1_b1, *gate1_b1;
  const float *spec1_b2, *shar1_b2, *gate1_b2;
  const float *tower_W, *tower_b;
  const _Float16 *w1f0, *w2f0, *w3f0, *w1f1, *w2f1, *w3f1;
  float* out;
};

__global__ __launch_bounds__(512) void ple_forward(PleParams P) {
  // LDS (~56 KB, well under the 320 KB WGP budget)
  __shared__ __align__(16) _Float16 xtile[16 * 96];        // embeddings, K-padded
  __shared__ __align__(16) _Float16 htile[16][16 * 64];    // per-unit hidden
  __shared__ __align__(16) float    otile[12][16 * 8];     // expert outputs (f32)
  __shared__ __align__(16) _Float16 gtile[4][16 * 32];     // gate hidden, K-padded
  __shared__ __align__(16) float    glog[4][16 * 16];      // gate logits
  __shared__ __align__(16) _Float16 tx1[3][16 * 32];       // level-1 task inputs
  __shared__ __align__(16) _Float16 sx1[16 * 32];          // level-1 shared input
  __shared__ __align__(16) float    tout[3][16 * 8];       // level-1 task outputs

  const int tid = threadIdx.x;
  const int w = tid >> 5, lane = tid & 31;
  const int hi = lane >> 4, ln = lane & 15;

  for (int tile = blockIdx.x; tile < NTILES; tile += gridDim.x) {
    const int r0 = tile * 16;

    // ---- Phase A: gather embeddings -> f16 tile [16 rows][96 K] (72 real) ----
    for (int i = tid; i < 16 * 96; i += 512) {
      const int r = i / 96, d = i % 96;
      float v = 0.0f;
      if (d < 72) {
        const int f = d >> 3, j = d & 7;
        const int idx = P.cat_idx[(r0 + r) * 9 + f];
        v = P.emb[(size_t)idx * 8 + j];
      }
      xtile[r * 96 + d] = (_Float16)v;
    }
    __syncthreads();

    // ---- Phase B: level-0 W1 (16 units, one wave each): [16x72]@[72x64] ----
    {
      v8f acc[4] = {{}, {}, {}, {}};
#pragma unroll
      for (int c = 0; c < 3; ++c) {
        v16h a = load_afrag(xtile, 96, c * 32, lane);
#pragma unroll
        for (int nt = 0; nt < 4; ++nt) {
          v16h b = load_bfrag(P.w1f0, w * 12 + c * 4 + nt, lane);
          acc[nt] = wmma_f16(a, b, acc[nt]);
        }
      }
      const float* b1 = (w < 9)  ? P.spec0_b1 + w * 64
                      : (w < 12) ? P.shar0_b1 + (w - 9) * 64
                      : (w < 15) ? P.gate0_b1 + (w - 12) * 64
                                 : P.sgate0_b1;
#pragma unroll
      for (int nt = 0; nt < 4; ++nt) {
        const float bv = b1[nt * 16 + ln];
#pragma unroll
        for (int v = 0; v < 8; ++v) {
          float x = acc[nt][v] + bv;
          x = x > 0.f ? x : 0.f;  // relu
          htile[w][(v + 8 * hi) * 64 + nt * 16 + ln] = (_Float16)x;
        }
      }
    }
    __syncthreads();

    // ---- Phase C: level-0 W2: [16x64]@[64x8] (+relu) ----
    {
      v8f acc = {};
#pragma unroll
      for (int c = 0; c < 2; ++c) {
        v16h a = load_afrag(htile[w], 64, c * 32, lane);
        v16h b = load_bfrag(P.w2f0, w * 2 + c, lane);
        acc = wmma_f16(a, b, acc);
      }
      const float* b2 = (w < 9)  ? P.spec0_b2 + w * 8
                      : (w < 12) ? P.shar0_b2 + (w - 9) * 8
                      : (w < 15) ? P.gate0_b2 + (w - 12) * 8
                                 : P.sgate0_b2;
      const float bv = (ln < 8) ? b2[ln] : 0.f;
      if (w < 12) {  // expert outputs
#pragma unroll
        for (int v = 0; v < 8; ++v) {
          float x = acc[v] + bv;
          x = x > 0.f ? x : 0.f;
          if (ln < 8) otile[w][(v + 8 * hi) * 8 + ln] = x;
        }
      } else {       // gate hidden -> K-padded f16 tile for the W3 WMMA
        const int g = w - 12;
#pragma unroll
        for (int v = 0; v < 8; ++v) {
          float x = acc[v] + bv;
          x = x > 0.f ? x : 0.f;
          const int row = v + 8 * hi;
          gtile[g][row * 32 + ln] = (_Float16)x;        // cols 8..15 are exact 0
          gtile[g][row * 32 + 16 + ln] = (_Float16)0.f; // K pad 16..31
        }
      }
    }
    __syncthreads();

    // ---- Phase C2: gate/sgate W3 logits: [16x8(pad32)]@[8x{6,12}] ----
    if (w >= 12) {
      const int g = w - 12;
      v16h a = load_afrag(gtile[g], 32, 0, lane);
      v16h b = load_bfrag(P.w3f0, g, lane);
      v8f z = {};
      v8f gl = wmma_f16(a, b, z);
#pragma unroll
      for (int v = 0; v < 8; ++v) glog[g][(v + 8 * hi) * 16 + ln] = gl[v];
    }
    __syncthreads();

    // ---- Phase D: level-0 softmax-gated mixing ----
    {
      const int u = tid >> 7, r = (tid >> 3) & 15, o = tid & 7;
      float val = 0.f;
      if (u < 3) {  // per-task gate over 3 spec + 3 shar
        float lg[6], mx = -1e30f;
#pragma unroll
        for (int e = 0; e < 6; ++e) { lg[e] = glog[u][r * 16 + e]; mx = fmaxf(mx, lg[e]); }
        float s = 0.f;
#pragma unroll
        for (int e = 0; e < 6; ++e) { lg[e] = __expf(lg[e] - mx); s += lg[e]; }
        const float inv = 1.f / s;
#pragma unroll
        for (int e = 0; e < 6; ++e) {
          const float* ex = (e < 3) ? otile[u * 3 + e] : otile[9 + (e - 3)];
          val += lg[e] * inv * ex[r * 8 + o];
        }
        tx1[u][r * 32 + o] = (_Float16)val;
      } else {      // shared gate over all 9 spec + 3 shar
        float lg[12], mx = -1e30f;
#pragma unroll
        for (int e = 0; e < 12; ++e) { lg[e] = glog[3][r * 16 + e]; mx = fmaxf(mx, lg[e]); }
        float s = 0.f;
#pragma unroll
        for (int e = 0; e < 12; ++e) { lg[e] = __expf(lg[e] - mx); s += lg[e]; }
        const float inv = 1.f / s;
#pragma unroll
        for (int e = 0; e < 12; ++e) val += lg[e] * inv * otile[e][r * 8 + o];
        sx1[r * 32 + o] = (_Float16)val;
      }
      // zero K-pad 8..31 of the level-1 inputs
      for (int i = tid; i < 4 * 16 * 24; i += 512) {
        const int uu = i / (16 * 24), rem = i % (16 * 24);
        const int rr = rem / 24, cc = 8 + rem % 24;
        _Float16* dst = (uu < 3) ? tx1[uu] : sx1;
        dst[rr * 32 + cc] = (_Float16)0.f;
      }
    }
    __syncthreads();

    // ---- Phase E: level-1 W1: [16x8(pad32)]@[8x64] (15 units) ----
    if (w < 15) {
      const _Float16* inx = (w < 9) ? tx1[w / 3] : (w < 12) ? sx1 : tx1[w - 12];
      v16h a = load_afrag(inx, 32, 0, lane);
      v8f acc[4] = {{}, {}, {}, {}};
#pragma unroll
      for (int nt = 0; nt < 4; ++nt) {
        v16h b = load_bfrag(P.w1f1, w * 4 + nt, lane);
        acc[nt] = wmma_f16(a, b, acc[nt]);
      }
      const float* b1 = (w < 9)  ? P.spec1_b1 + w * 64
                      : (w < 12) ? P.shar1_b1 + (w - 9) * 64
                                 : P.gate1_b1 + (w - 12) * 64;
#pragma unroll
      for (int nt = 0; nt < 4; ++nt) {
        const float bv = b1[nt * 16 + ln];
#pragma unroll
        for (int v = 0; v < 8; ++v) {
          float x = acc[nt][v] + bv;
          x = x > 0.f ? x : 0.f;
          htile[w][(v + 8 * hi) * 64 + nt * 16 + ln] = (_Float16)x;
        }
      }
    }
    __syncthreads();

    // ---- Phase F: level-1 W2: [16x64]@[64x8] ----
    if (w < 15) {
      v8f acc = {};
#pragma unroll
      for (int c = 0; c < 2; ++c) {
        v16h a = load_afrag(htile[w], 64, c * 32, lane);
        v16h b = load_bfrag(P.w2f1, w * 2 + c, lane);
        acc = wmma_f16(a, b, acc);
      }
      const float* b2 = (w < 9)  ? P.spec1_b2 + w * 8
                      : (w < 12) ? P.shar1_b2 + (w - 9) * 8
                                 : P.gate1_b2 + (w - 12) * 8;
      const float bv = (ln < 8) ? b2[ln] : 0.f;
      if (w < 12) {
#pragma unroll
        for (int v = 0; v < 8; ++v) {
          float x = acc[v] + bv;
          x = x > 0.f ? x : 0.f;
          if (ln < 8) otile[w][(v + 8 * hi) * 8 + ln] = x;
        }
      } else {
        const int g = w - 12;
#pragma unroll
        for (int v = 0; v < 8; ++v) {
          float x = acc[v] + bv;
          x = x > 0.f ? x : 0.f;
          const int row = v + 8 * hi;
          gtile[g][row * 32 + ln] = (_Float16)x;
          gtile[g][row * 32 + 16 + ln] = (_Float16)0.f;
        }
      }
    }
    __syncthreads();

    // ---- Phase G: gate1 W3 logits ----
    if (w >= 12 && w < 15) {
      const int g = w - 12;
      v16h a = load_afrag(gtile[g], 32, 0, lane);
      v16h b = load_bfrag(P.w3f1, g, lane);
      v8f z = {};
      v8f gl = wmma_f16(a, b, z);
#pragma unroll
      for (int v = 0; v < 8; ++v) glog[g][(v + 8 * hi) * 16 + ln] = gl[v];
    }
    __syncthreads();

    // ---- Phase H: level-1 mixing ----
    if (tid < 384) {
      const int t = tid >> 7, r = (tid >> 3) & 15, o = tid & 7;
      float lg[6], mx = -1e30f;
#pragma unroll
      for (int e = 0; e < 6; ++e) { lg[e] = glog[t][r * 16 + e]; mx = fmaxf(mx, lg[e]); }
      float s = 0.f;
#pragma unroll
      for (int e = 0; e < 6; ++e) { lg[e] = __expf(lg[e] - mx); s += lg[e]; }
      const float inv = 1.f / s;
      float val = 0.f;
#pragma unroll
      for (int e = 0; e < 6; ++e) {
        const float* ex = (e < 3) ? otile[t * 3 + e] : otile[9 + (e - 3)];
        val += lg[e] * inv * ex[r * 8 + o];
      }
      tout[t][r * 8 + o] = val;
    }
    __syncthreads();

    // ---- Towers: sigmoid(task_x @ tower_W + tower_b) -> out[t*B + b] ----
    if (tid < 48) {
      const int t = tid / 16, r = tid % 16;
      float s = P.tower_b[t];
#pragma unroll
      for (int o = 0; o < 8; ++o) s += tout[t][r * 8 + o] * P.tower_W[t * 8 + o];
      P.out[(size_t)t * BATCH + r0 + r] = 1.f / (1.f + __expf(-s));
    }
    __syncthreads();  // LDS reuse fence before next tile
  }
}

// ---------------------------------------------------------------- launch
extern "C" void kernel_launch(void* const* d_in, const int* in_sizes, int n_in,
                              void* d_out, int out_size, void* d_ws, size_t ws_size,
                              hipStream_t stream) {
  (void)in_sizes; (void)n_in; (void)out_size; (void)ws_size;
  _Float16* ws = (_Float16*)d_ws;

  auto PK = [&](const void* W, size_t dstOffHalves, int K, int N, int nC, int nT,
                int units) {
    const int total = nC * nT * 512 * units;
    const int blocks = (total + 255) / 256;
    hipLaunchKernelGGL(pack_bfrag, dim3(blocks), dim3(256), 0, stream,
                       (const float*)W, ws + dstOffHalves, K, N, nC, nT, units);
  };

  // Level-0 W1 (72->64): units packed in order spec(9), shar(3), gate(3), sgate(1)
  PK(d_in[1],  OFF_W1F0 + 0 * 6144, 72, 64, 3, 4, 9);
  PK(d_in[5],  OFF_W1F0 + 9 * 6144, 72, 64, 3, 4, 3);
  PK(d_in[9],  OFF_W1F0 + 12 * 6144, 72, 64, 3, 4, 3);
  PK(d_in[14], OFF_W1F0 + 15 * 6144, 72, 64, 3, 4, 1);
  // Level-0 W2 (64->8)
  PK(d_in[3],  OFF_W2F0 + 0 * 1024, 64, 8, 2, 1, 9);
  PK(d_in[7],  OFF_W2F0 + 9 * 1024, 64, 8, 2, 1, 3);
  PK(d_in[11], OFF_W2F0 + 12 * 1024, 64, 8, 2, 1, 3);
  PK(d_in[16], OFF_W2F0 + 15 * 1024, 64, 8, 2, 1, 1);
  // Level-0 gate W3 (8->6) and sgate W3 (8->12)
  PK(d_in[13], OFF_W3F0 + 0 * 512, 8, 6, 1, 1, 3);
  PK(d_in[18], OFF_W3F0 + 3 * 512, 8, 12, 1, 1, 1);
  // Level-1 W1 (8->64): spec(9), shar(3), gate(3)
  PK(d_in[19], OFF_W1F1 + 0 * 2048, 8, 64, 1, 4, 9);
  PK(d_in[23], OFF_W1F1 + 9 * 2048, 8, 64, 1, 4, 3);
  PK(d_in[27], OFF_W1F1 + 12 * 2048, 8, 64, 1, 4, 3);
  // Level-1 W2 (64->8)
  PK(d_in[21], OFF_W2F1 + 0 * 1024, 64, 8, 2, 1, 9);
  PK(d_in[25], OFF_W2F1 + 9 * 1024, 64, 8, 2, 1, 3);
  PK(d_in[29], OFF_W2F1 + 12 * 1024, 64, 8, 2, 1, 3);
  // Level-1 gate W3 (8->6)
  PK(d_in[31], OFF_W3F1 + 0 * 512, 8, 6, 1, 1, 3);

  PleParams P;
  P.emb = (const float*)d_in[0];
  P.spec0_b1 = (const float*)d_in[2];  P.spec0_b2 = (const float*)d_in[4];
  P.shar0_b1 = (const float*)d_in[6];  P.shar0_b2 = (const float*)d_in[8];
  P.gate0_b1 = (const float*)d_in[10]; P.gate0_b2 = (const float*)d_in[12];
  P.sgate0_b1 = (const float*)d_in[15]; P.sgate0_b2 = (const float*)d_in[17];
  P.spec1_b1 = (const float*)d_in[20]; P.spec1_b2 = (const float*)d_in[22];
  P.shar1_b1 = (const float*)d_in[24]; P.shar1_b2 = (const float*)d_in[26];
  P.gate1_b1 = (const float*)d_in[28]; P.gate1_b2 = (const float*)d_in[30];
  P.tower_W = (const float*)d_in[32]; P.tower_b = (const float*)d_in[33];
  P.cat_idx = (const int*)d_in[34];
  P.w1f0 = ws + OFF_W1F0; P.w2f0 = ws + OFF_W2F0; P.w3f0 = ws + OFF_W3F0;
  P.w1f1 = ws + OFF_W1F1; P.w2f1 = ws + OFF_W2F1; P.w3f1 = ws + OFF_W3F1;
  P.out = (float*)d_out;

  hipLaunchKernelGGL(ple_forward, dim3(NTILES), dim3(512), 0, stream, P);
}
